// LConvBlock_16930761081218
// MI455X (gfx1250) — matrix operand order repaired and checked by reading, not compiled, not run on hardware
//
#include <hip/hip_runtime.h>
#include <math.h>

// ---------------------------------------------------------------------------
// LConv block for MI455X (gfx1250).
//  - bf16 WMMA (v_wmma_f32_16x16x32_bf16) for all GEMMs, f32 accumulate
//  - async global->LDS (GLOBAL_LOAD_ASYNC_TO_LDS_B128) double-buffered k-loops
//  - fused GLU / conv+residual / ReLU / residual+LayerNorm epilogues
//   B=16 T=2048 C=512 H=8 K=31, M = B*T = 32768
// ---------------------------------------------------------------------------

typedef __attribute__((ext_vector_type(16))) __bf16 v16bf;
typedef __attribute__((ext_vector_type(8)))  float  v8f;

#define Bdim 16
#define Tdim 2048
#define Cdim 512
#define Hdim 8
#define Kdim 31
#define Mrows (Bdim * Tdim)          // 32768
#define TCdim (Tdim * Cdim)          // 2^20
#define LSTR 40                      // LDS row stride (bf16): 80B, 16B-aligned

// ---- CDNA5 async copy: 16 bytes per lane, global -> LDS (ASYNCcnt) --------
__device__ __forceinline__ void async_b128(void* lds, const void* g) {
  unsigned l = (unsigned)(unsigned long long)lds;        // addr[31:0] = LDS offset
  unsigned long long a = (unsigned long long)g;
  asm volatile("global_load_async_to_lds_b128 %0, %1, off"
               :: "v"(l), "v"(a) : "memory");
}
#define WAIT_ASYNC(n) asm volatile("s_wait_asynccnt %0" :: "i"(n) : "memory")

// ---- wave32 WMMA fragment loaders (CDNA5 ISA 7.12.2 layouts) --------------
// A (16-bit, 16x32 MxK): lane&15 = row M; lane>=16 -> kbase=8.
// elements 0..7 = K kbase..kbase+7 ; elements 8..15 = K 16+kbase..16+kbase+7
__device__ __forceinline__ v16bf ld_afrag(const __bf16* As, int mbase, int lane) {
  const __bf16* p = As + (size_t)(mbase + (lane & 15)) * LSTR + ((lane >> 4) << 3);
  v16bf a;
#pragma unroll
  for (int i = 0; i < 8; ++i) { a[i] = p[i]; a[8 + i] = p[16 + i]; }
  return a;
}
// B (16-bit, 32x16 KxN), LDS holds B transposed [n][k]:
// lane&15 = N; lane>=16 -> kbase=8; e0..7 = K kbase+e ; e8..15 = K 16+kbase+(e-8)
__device__ __forceinline__ v16bf ld_bfrag(const __bf16* Bs, int nbase, int lane) {
  const __bf16* p = Bs + (size_t)(nbase + (lane & 15)) * LSTR + ((lane >> 4) << 3);
  v16bf b;
#pragma unroll
  for (int i = 0; i < 8; ++i) { b[i] = p[i]; b[8 + i] = p[16 + i]; }
  return b;
}

// ---------------------------------------------------------------------------
// Prep kernels: f32 -> bf16, and f32 [K][N] -> bf16 [N][K] (transpose+convert)
// ---------------------------------------------------------------------------
__global__ __launch_bounds__(256) void k_f32_to_bf16(
    const float* __restrict__ in, __bf16* __restrict__ out, int n) {
  int i = blockIdx.x * 256 + threadIdx.x;
  if (i < n) out[i] = (__bf16)in[i];
}

__global__ __launch_bounds__(256) void k_transpose_cvt(
    const float* __restrict__ in, __bf16* __restrict__ out, int K, int N) {
  __shared__ float tile[32][33];
  const int k0 = blockIdx.y * 32, n0 = blockIdx.x * 32;
  const int c = threadIdx.x & 31, r0 = threadIdx.x >> 5;
#pragma unroll
  for (int r = r0; r < 32; r += 8)
    tile[r][c] = in[(size_t)(k0 + r) * N + n0 + c];
  __syncthreads();
#pragma unroll
  for (int r = r0; r < 32; r += 8)
    out[(size_t)(n0 + r) * K + k0 + c] = (__bf16)tile[c][r];
}

// ---------------------------------------------------------------------------
// Kernel 1: h = GLU(x @ W_act + b_act), masked, stored bf16.
// 128(M) x 64(N) tile, 8 waves = 4(M) x 2(N). act & gate B-streams in one loop.
// Double-buffered async tiles: 4 async b128 per lane per k-step.
// ---------------------------------------------------------------------------
__global__ __launch_bounds__(256) void k_gemm1_glu(
    const __bf16* __restrict__ xb, const unsigned char* __restrict__ mask,
    const __bf16* __restrict__ WaT, const float* __restrict__ bias,
    __bf16* __restrict__ hout) {
  __shared__ __bf16 As[2][128][LSTR];
  __shared__ __bf16 Ba[2][64][LSTR];
  __shared__ __bf16 Bg[2][64][LSTR];
  const int tid = threadIdx.x, lane = tid & 31, wave = tid >> 5;
  const int wm = wave & 3, wn = wave >> 2;
  const int m0 = blockIdx.x * 128, n0 = blockIdx.y * 64;
  const int ar = tid >> 2, ac = (tid & 3) * 8;   // 64 rows x 4 x 16B chunks
  v8f aa[2][2] = {}; v8f ag[2][2] = {};

  auto issue = [&](int kb, int buf) {
    const int k0 = kb * 32;
    async_b128(&As[buf][ar][ac],      xb  + (size_t)(m0 + ar) * Cdim + k0 + ac);
    async_b128(&As[buf][64 + ar][ac], xb  + (size_t)(m0 + 64 + ar) * Cdim + k0 + ac);
    async_b128(&Ba[buf][ar][ac],      WaT + (size_t)(n0 + ar) * Cdim + k0 + ac);
    async_b128(&Bg[buf][ar][ac],      WaT + (size_t)(n0 + Cdim + ar) * Cdim + k0 + ac);
  };

  const int NK = Cdim / 32;   // 16
  issue(0, 0);
  issue(1, 1);
  for (int kb = 0; kb < NK; ++kb) {
    const int buf = kb & 1;
    if (kb + 1 < NK) { WAIT_ASYNC(4); } else { WAIT_ASYNC(0); }
    __syncthreads();
    v16bf af[2], ba[2], bg[2];
#pragma unroll
    for (int mt = 0; mt < 2; ++mt) af[mt] = ld_afrag(&As[buf][0][0], wm * 32 + mt * 16, lane);
#pragma unroll
    for (int nt = 0; nt < 2; ++nt) {
      ba[nt] = ld_bfrag(&Ba[buf][0][0], wn * 32 + nt * 16, lane);
      bg[nt] = ld_bfrag(&Bg[buf][0][0], wn * 32 + nt * 16, lane);
    }
#pragma unroll
    for (int mt = 0; mt < 2; ++mt)
#pragma unroll
      for (int nt = 0; nt < 2; ++nt) {
        aa[mt][nt] = __builtin_amdgcn_wmma_f32_16x16x32_bf16(
            false, af[mt], false, ba[nt], (short)0, aa[mt][nt], false, false);
        ag[mt][nt] = __builtin_amdgcn_wmma_f32_16x16x32_bf16(
            false, af[mt], false, bg[nt], (short)0, ag[mt][nt], false, false);
      }
    __syncthreads();
    if (kb + 2 < NK) issue(kb + 2, buf);
  }

  const int ci = lane & 15, hi = lane >> 4;
#pragma unroll
  for (int mt = 0; mt < 2; ++mt)
#pragma unroll
    for (int nt = 0; nt < 2; ++nt)
#pragma unroll
      for (int j = 0; j < 8; ++j) {
        int row = m0 + wm * 32 + mt * 16 + hi * 8 + j;
        int col = n0 + wn * 32 + nt * 16 + ci;
        float a = aa[mt][nt][j] + bias[col];
        float g = ag[mt][nt][j] + bias[col + Cdim];
        float v = a / (1.0f + __expf(-g));
        if (mask[row]) v = 0.0f;
        hout[(size_t)row * Cdim + col] = (__bf16)v;
      }
}

// ---------------------------------------------------------------------------
// Kernel 2: depthwise conv (softmax kernel per head) + residual.
// y = x + conv (f32 for LN-residual), yb = bf16 copy (GEMM2 A operand).
// h (32MB bf16) is L2-resident; taps read straight from L2.
// ---------------------------------------------------------------------------
__global__ __launch_bounds__(256) void k_conv(
    const __bf16* __restrict__ h, const float* __restrict__ x,
    const float* __restrict__ cw, float* __restrict__ y,
    __bf16* __restrict__ yb) {
  __shared__ float sw[Hdim][32];
  const int tid = threadIdx.x;
  if (tid < Hdim) {
    float mx = cw[tid * Kdim];
    for (int k = 1; k < Kdim; ++k) mx = fmaxf(mx, cw[tid * Kdim + k]);
    float s = 0.f;
    for (int k = 0; k < Kdim; ++k) {
      float e = __expf(cw[tid * Kdim + k] - mx);
      sw[tid][k] = e; s += e;
    }
    float inv = 1.f / s;
    for (int k = 0; k < Kdim; ++k) sw[tid][k] *= inv;
  }
  __syncthreads();
  size_t idx = (size_t)blockIdx.x * 256 + tid;
  int c = (int)(idx & (Cdim - 1));
  int t = (int)((idx >> 9) & (Tdim - 1));
  size_t base = (idx & ~(size_t)(TCdim - 1)) + c;   // (b, t=0, c)
  const int head = c >> 6;                          // C/H = 64
  float sum = 0.f;
#pragma unroll
  for (int k = 0; k < Kdim; ++k) {
    int tt = t + k - (Kdim / 2);
    if (tt >= 0 && tt < Tdim)
      sum += sw[head][k] * (float)h[base + (size_t)tt * Cdim];
  }
  float v = x[idx] + sum;
  y[idx] = v;
  yb[idx] = (__bf16)v;
}

// ---------------------------------------------------------------------------
// Kernel 3: t = relu(y @ W_fc1 + b_fc1), stored bf16. 128x64 tile, async x2 buf.
// ---------------------------------------------------------------------------
__global__ __launch_bounds__(256) void k_gemm2_relu(
    const __bf16* __restrict__ yb, const __bf16* __restrict__ W1T,
    const float* __restrict__ bias, __bf16* __restrict__ tout) {
  __shared__ __bf16 As[2][128][LSTR];
  __shared__ __bf16 Bs[2][64][LSTR];
  const int tid = threadIdx.x, lane = tid & 31, wave = tid >> 5;
  const int wm = wave & 3, wn = wave >> 2;
  const int m0 = blockIdx.x * 128, n0 = blockIdx.y * 64;
  const int ar = tid >> 2, ac = (tid & 3) * 8;
  v8f acc[2][2] = {};

  auto issue = [&](int kb, int buf) {
    const int k0 = kb * 32;
    async_b128(&As[buf][ar][ac],      yb  + (size_t)(m0 + ar) * Cdim + k0 + ac);
    async_b128(&As[buf][64 + ar][ac], yb  + (size_t)(m0 + 64 + ar) * Cdim + k0 + ac);
    async_b128(&Bs[buf][ar][ac],      W1T + (size_t)(n0 + ar) * Cdim + k0 + ac);
  };

  const int NK = Cdim / 32;   // 16
  issue(0, 0);
  issue(1, 1);
  for (int kb = 0; kb < NK; ++kb) {
    const int buf = kb & 1;
    if (kb + 1 < NK) { WAIT_ASYNC(3); } else { WAIT_ASYNC(0); }
    __syncthreads();
    v16bf af[2], bf[2];
#pragma unroll
    for (int mt = 0; mt < 2; ++mt) af[mt] = ld_afrag(&As[buf][0][0], wm * 32 + mt * 16, lane);
#pragma unroll
    for (int nt = 0; nt < 2; ++nt) bf[nt] = ld_bfrag(&Bs[buf][0][0], wn * 32 + nt * 16, lane);
#pragma unroll
    for (int mt = 0; mt < 2; ++mt)
#pragma unroll
      for (int nt = 0; nt < 2; ++nt)
        acc[mt][nt] = __builtin_amdgcn_wmma_f32_16x16x32_bf16(
            false, af[mt], false, bf[nt], (short)0, acc[mt][nt], false, false);
    __syncthreads();
    if (kb + 2 < NK) issue(kb + 2, buf);
  }

  const int ci = lane & 15, hi = lane >> 4;
#pragma unroll
  for (int mt = 0; mt < 2; ++mt)
#pragma unroll
    for (int nt = 0; nt < 2; ++nt)
#pragma unroll
      for (int j = 0; j < 8; ++j) {
        int row = m0 + wm * 32 + mt * 16 + hi * 8 + j;
        int col = n0 + wn * 32 + nt * 16 + ci;
        float v = fmaxf(acc[mt][nt][j] + bias[col], 0.0f);
        tout[(size_t)row * (4 * Cdim) + col] = (__bf16)v;
      }
}

// ---------------------------------------------------------------------------
// Kernel 4: out = LN(y + t @ W_fc2 + b_fc2) * g + b, masked.
// Block = 64 rows x full 512 cols (LN fused). 8 waves split N, 4x4 tiles/wave.
// Fully async double-buffered (9 async b128 per lane per k-step, K=2048).
// ---------------------------------------------------------------------------
__global__ __launch_bounds__(256) void k_gemm3_ln(
    const __bf16* __restrict__ tp, const __bf16* __restrict__ W2T,
    const float* __restrict__ bias, const float* __restrict__ resid,
    const float* __restrict__ gamma, const float* __restrict__ beta,
    const unsigned char* __restrict__ mask, float* __restrict__ out) {
  __shared__ __bf16 As[2][64][LSTR];
  __shared__ __bf16 Bs[2][512][LSTR];
  __shared__ float psum[8][64];
  __shared__ float psq[8][64];
  __shared__ float smean[64], srstd[64];
  const int tid = threadIdx.x, lane = tid & 31, wn = tid >> 5;
  const int m0 = blockIdx.x * 64;
  const int ar = tid >> 2, ac = (tid & 3) * 8;
  v8f acc[4][4] = {};

  auto issue = [&](int kb, int buf) {
    const int k0 = kb * 32;
    async_b128(&As[buf][ar][ac], tp + (size_t)(m0 + ar) * (4 * Cdim) + k0 + ac);
#pragma unroll
    for (int i = 0; i < 8; ++i)
      async_b128(&Bs[buf][ar + 64 * i][ac],
                 W2T + (size_t)(ar + 64 * i) * (4 * Cdim) + k0 + ac);
  };

  const int NK = (4 * Cdim) / 32;   // 64
  issue(0, 0);
  issue(1, 1);
  for (int kb = 0; kb < NK; ++kb) {
    const int buf = kb & 1;
    if (kb + 1 < NK) { WAIT_ASYNC(9); } else { WAIT_ASYNC(0); }
    __syncthreads();
    v16bf af[4], bf[4];
#pragma unroll
    for (int mt = 0; mt < 4; ++mt) af[mt] = ld_afrag(&As[buf][0][0], mt * 16, lane);
#pragma unroll
    for (int nt = 0; nt < 4; ++nt) bf[nt] = ld_bfrag(&Bs[buf][0][0], wn * 64 + nt * 16, lane);
#pragma unroll
    for (int mt = 0; mt < 4; ++mt)
#pragma unroll
      for (int nt = 0; nt < 4; ++nt)
        acc[mt][nt] = __builtin_amdgcn_wmma_f32_16x16x32_bf16(
            false, af[mt], false, bf[nt], (short)0, acc[mt][nt], false, false);
    __syncthreads();
    if (kb + 2 < NK) issue(kb + 2, buf);
  }

  const int ci = lane & 15, hi = lane >> 4;
  // bias + residual into registers; per-row partial sums over this wave's 64 cols
#pragma unroll
  for (int mt = 0; mt < 4; ++mt) {
#pragma unroll
    for (int j = 0; j < 8; ++j) {
      float v = 0.f, v2 = 0.f;
      int rl = mt * 16 + hi * 8 + j;
      int row = m0 + rl;
#pragma unroll
      for (int nt = 0; nt < 4; ++nt) {
        int col = wn * 64 + nt * 16 + ci;
        float u = acc[mt][nt][j] + bias[col] + resid[(size_t)row * Cdim + col];
        acc[mt][nt][j] = u;
        v += u; v2 += u * u;
      }
      // reduce across the 16 lanes of this column half (xor masks stay in-half)
#pragma unroll
      for (int off = 1; off < 16; off <<= 1) {
        v  += __shfl_xor(v,  off, 32);
        v2 += __shfl_xor(v2, off, 32);
      }
      if (ci == 0) { psum[wn][rl] = v; psq[wn][rl] = v2; }
    }
  }
  __syncthreads();
  if (tid < 64) {
    float s = 0.f, s2 = 0.f;
#pragma unroll
    for (int w = 0; w < 8; ++w) { s += psum[w][tid]; s2 += psq[w][tid]; }
    float mu  = s * (1.0f / Cdim);
    float var = s2 * (1.0f / Cdim) - mu * mu;
    smean[tid] = mu;
    srstd[tid] = rsqrtf(var + 1e-5f);
  }
  __syncthreads();
#pragma unroll
  for (int mt = 0; mt < 4; ++mt)
#pragma unroll
    for (int nt = 0; nt < 4; ++nt)
#pragma unroll
      for (int j = 0; j < 8; ++j) {
        int rl = mt * 16 + hi * 8 + j;
        int row = m0 + rl;
        int col = wn * 64 + nt * 16 + ci;
        float o = (acc[mt][nt][j] - smean[rl]) * srstd[rl] * gamma[col] + beta[col];
        if (mask[row]) o = 0.f;
        out[(size_t)row * Cdim + col] = o;
      }
}

// ---------------------------------------------------------------------------
extern "C" void kernel_launch(void* const* d_in, const int* in_sizes, int n_in,
                              void* d_out, int out_size, void* d_ws, size_t ws_size,
                              hipStream_t stream) {
  (void)in_sizes; (void)n_in; (void)out_size; (void)ws_size;
  const float*         x      = (const float*)d_in[0];
  const unsigned char* mask   = (const unsigned char*)d_in[1];   // jnp bool = 1 byte
  const float*         W_act  = (const float*)d_in[2];
  const float*         b_act  = (const float*)d_in[3];
  const float*         conv_w = (const float*)d_in[4];
  const float*         W_fc1  = (const float*)d_in[5];
  const float*         b_fc1  = (const float*)d_in[6];
  const float*         W_fc2  = (const float*)d_in[7];
  const float*         b_fc2  = (const float*)d_in[8];
  const float*         ln_g   = (const float*)d_in[9];
  const float*         ln_b   = (const float*)d_in[10];
  float* out = (float*)d_out;

  // Workspace layout (region [0,128MB) is reused: xb+h first, then t):
  char* ws = (char*)d_ws;
  const size_t MB32 = (size_t)Mrows * Cdim * 2;   // 32MB (bf16 plane)
  __bf16* xb  = (__bf16*)(ws);                    // [0,32MB)   dead after gemm1
  __bf16* h   = (__bf16*)(ws + MB32);             // [32,64MB)  dead after conv
  __bf16* t   = (__bf16*)(ws);                    // [0,128MB)  written by gemm2
  float*  y   = (float*) (ws + 4 * MB32);         // [128,192MB)
  __bf16* yb  = (__bf16*)(ws + 6 * MB32);         // [192,224MB)
  __bf16* WaT = (__bf16*)(ws + 7 * MB32);                         // 1MB  (1024x512)
  __bf16* W1T = (__bf16*)(ws + 7 * MB32 + (size_t)1024 * 512 * 2);// 2MB  (2048x512)
  __bf16* W2T = (__bf16*)(ws + 7 * MB32 + (size_t)3072 * 512 * 2);// 2MB  (512x2048)

  // --- operand prep: bf16 conversion + weight transposes -------------------
  k_f32_to_bf16<<<(Mrows * Cdim) / 256, 256, 0, stream>>>(x, xb, Mrows * Cdim);
  k_transpose_cvt<<<dim3(1024 / 32, 512 / 32), 256, 0, stream>>>(W_act, WaT, 512, 1024);
  k_transpose_cvt<<<dim3(2048 / 32, 512 / 32), 256, 0, stream>>>(W_fc1, W1T, 512, 2048);
  k_transpose_cvt<<<dim3(512 / 32, 2048 / 32), 256, 0, stream>>>(W_fc2, W2T, 2048, 512);

  dim3 g1(Mrows / 128, Cdim / 64);            // 256 x 8
  k_gemm1_glu<<<g1, 256, 0, stream>>>(xb, mask, WaT, b_act, h);

  k_conv<<<(Bdim * Tdim * Cdim) / 256, 256, 0, stream>>>(h, x, conv_w, y, yb);

  dim3 g2(Mrows / 128, (4 * Cdim) / 64);      // 256 x 32
  k_gemm2_relu<<<g2, 256, 0, stream>>>(yb, W1T, b_fc1, t);

  dim3 g3(Mrows / 64, 1);                     // 512
  k_gemm3_ln<<<g3, 256, 0, stream>>>(t, W2T, b_fc2, y, ln_g, ln_b, mask, out);
}
